// SelfAttentionHead_31035433680994
// MI455X (gfx1250) — matrix operand order repaired and checked by reading, not compiled, not run on hardware
//
#include <hip/hip_runtime.h>

// ---------------------------------------------------------------------------
// Types matching CDNA5 WMMA builtin signatures
// ---------------------------------------------------------------------------
typedef __attribute__((ext_vector_type(16))) __bf16 v16bf;
typedef __attribute__((ext_vector_type(8)))  __bf16 v8bf;
typedef __attribute__((ext_vector_type(8)))  float  v8f;
typedef __attribute__((ext_vector_type(4)))  float  v4f;

#define NEG_BIG (-3.0e38f)

// sched_group_barrier masks
#define SG_VALU  0x002
#define SG_WMMA  0x008
#define SG_VMEMR 0x020
#define SG_DSR   0x100

static __device__ __forceinline__ __bf16 f2bf(float f) {
  return (__bf16)f;   // native v_cvt_*bf16_f32
}

static __device__ __forceinline__ v16bf mk16(v8bf lo, v8bf hi) {
  v16bf r;
#pragma unroll
  for (int i = 0; i < 8; ++i) { r[i] = lo[i]; r[i + 8] = hi[i]; }
  return r;
}

static __device__ __forceinline__ v8bf ld8(const __bf16* p) {
  return *(const v8bf*)p;
}

static __device__ __forceinline__ v8f zero8() {
  v8f v = {0.f, 0.f, 0.f, 0.f, 0.f, 0.f, 0.f, 0.f};
  return v;
}

static __device__ __forceinline__ v8f wmma_bf16(v16bf a, v16bf b, v8f c) {
  return __builtin_amdgcn_wmma_f32_16x16x32_bf16(false, a, false, b, (short)0,
                                                 c, false, false);
}

// Problem constants
#define B_  256
#define T_  256
#define DM_ 384
#define DK_ 64
#define DV_ 64
#define NCOLS 192   // q(64) | k(64) | v(64)

// ---------------------------------------------------------------------------
// Kernel 0: pack W_q|W_k|W_v into bf16 Wt[192][384] (output-col major,
// input-channel contiguous) so WMMA B-fragments are contiguous 16-B loads.
// ---------------------------------------------------------------------------
__global__ void wt_pack_kernel(const float* __restrict__ Wq,
                               const float* __restrict__ Wk,
                               const float* __restrict__ Wv,
                               __bf16* __restrict__ Wt) {
  int idx = blockIdx.x * 256 + threadIdx.x;
  if (idx >= NCOLS * DM_) return;
  int n = idx / DM_;          // output column 0..191
  int k = idx % DM_;          // input channel
  const float* W = (n < 64) ? Wq : (n < 128) ? Wk : Wv;
  Wt[idx] = f2bf(W[k * 64 + (n & 63)]);
}

// ---------------------------------------------------------------------------
// Kernel 1: QKV projection. One wave computes a 16-row tile of q,k,v.
// Software-pipelined: B-fragment group g+1 is fetched before the WMMAs of
// group g, and sched_group_barriers pin VMEM-before-WMMA ordering.
// ---------------------------------------------------------------------------
__global__ void __launch_bounds__(256, 1)
qkv_proj_kernel(const float* __restrict__ x, const __bf16* __restrict__ Wt,
                __bf16* __restrict__ qb, __bf16* __restrict__ kb,
                __bf16* __restrict__ vT) {
  const int lane = threadIdx.x & 31;
  const int wv   = threadIdx.x >> 5;
  const int tile = blockIdx.x * 8 + wv;        // 4096 tiles total
  const int b    = tile >> 4;                  // T/16 = 16 tiles per batch
  const int trow = (tile & 15) << 4;
  const int l15  = lane & 15;
  const int h    = lane >> 4;

  v8f acc[12];
#pragma unroll
  for (int i = 0; i < 12; ++i) acc[i] = zero8();

  // A-matrix: lane holds row M = l15
  const float* xrow = x + (size_t)(b * T_ + trow + l15) * DM_;
  // B-fragment base for this lane: row = output column, K contiguous
  const __bf16* wbase = Wt + (size_t)l15 * DM_;

  for (int kk = 0; kk < 12; ++kk) {            // K = 32 per step
    // A fragment: K = 32kk + 16h + [0..15]
    const int ab = kk * 32 + 16 * h;
    v4f f0 = *(const v4f*)(xrow + ab);
    v4f f1 = *(const v4f*)(xrow + ab + 4);
    v4f g0 = *(const v4f*)(xrow + ab + 8);
    v4f g1 = *(const v4f*)(xrow + ab + 12);
    v16bf a;
#pragma unroll
    for (int i = 0; i < 4; ++i) {
      a[i]      = f2bf(f0[i]);
      a[i + 4]  = f2bf(f1[i]);
      a[i + 8]  = f2bf(g0[i]);
      a[i + 12] = f2bf(g1[i]);
    }
    const __bf16* wk = wbase + kk * 32 + 16 * h;

    // prefetch group 0
    v16bf nb0, nb1, nb2, nb3;
    {
      const __bf16* w0 = wk;
      const __bf16* w1 = wk + (size_t)1 * 16 * DM_;
      const __bf16* w2 = wk + (size_t)2 * 16 * DM_;
      const __bf16* w3 = wk + (size_t)3 * 16 * DM_;
      nb0 = mk16(ld8(w0), ld8(w0 + 8));
      nb1 = mk16(ld8(w1), ld8(w1 + 8));
      nb2 = mk16(ld8(w2), ld8(w2 + 8));
      nb3 = mk16(ld8(w3), ld8(w3 + 8));
    }
#pragma unroll
    for (int g = 0; g < 3; ++g) {
      v16bf b0 = nb0, b1 = nb1, b2 = nb2, b3 = nb3;
      if (g < 2) {  // fetch next group before consuming current one
        const __bf16* w0 = wk + (size_t)(4 * g + 4) * 16 * DM_;
        const __bf16* w1 = wk + (size_t)(4 * g + 5) * 16 * DM_;
        const __bf16* w2 = wk + (size_t)(4 * g + 6) * 16 * DM_;
        const __bf16* w3 = wk + (size_t)(4 * g + 7) * 16 * DM_;
        nb0 = mk16(ld8(w0), ld8(w0 + 8));
        nb1 = mk16(ld8(w1), ld8(w1 + 8));
        nb2 = mk16(ld8(w2), ld8(w2 + 8));
        nb3 = mk16(ld8(w3), ld8(w3 + 8));
      }
      acc[4 * g + 0] = wmma_bf16(a, b0, acc[4 * g + 0]);
      acc[4 * g + 1] = wmma_bf16(a, b1, acc[4 * g + 1]);
      acc[4 * g + 2] = wmma_bf16(a, b2, acc[4 * g + 2]);
      acc[4 * g + 3] = wmma_bf16(a, b3, acc[4 * g + 3]);
    }
    // Pin schedule: A loads + B groups 0,1 (20 vmem reads) ahead of first
    // WMMA quad; then alternate B group 2 with remaining WMMA quads.
    __builtin_amdgcn_sched_group_barrier(SG_VMEMR, 12, 0); // A(4) + Bg0(8)
    __builtin_amdgcn_sched_group_barrier(SG_VALU,   8, 0); // cvt_pk_bf16
    __builtin_amdgcn_sched_group_barrier(SG_VMEMR,  8, 0); // Bg1
    __builtin_amdgcn_sched_group_barrier(SG_WMMA,   4, 0); // wmma g0
    __builtin_amdgcn_sched_group_barrier(SG_VMEMR,  8, 0); // Bg2
    __builtin_amdgcn_sched_group_barrier(SG_WMMA,   4, 0); // wmma g1
    __builtin_amdgcn_sched_group_barrier(SG_WMMA,   4, 0); // wmma g2
  }

  // C/D layout: VGPR r -> M = r + 8h, N = l15
#pragma unroll
  for (int nt = 0; nt < 12; ++nt) {
#pragma unroll
    for (int r = 0; r < 8; ++r) {
      const int t   = trow + r + 8 * h;
      const int col = nt * 16 + l15;
      __bf16 v = f2bf(acc[nt][r]);
      if (col < 64)
        qb[(size_t)(b * T_ + t) * DK_ + col] = v;
      else if (col < 128)
        kb[(size_t)(b * T_ + t) * DK_ + (col - 64)] = v;
      else
        vT[(size_t)b * DV_ * T_ + (size_t)(col - 128) * T_ + t] = v;
    }
  }
}

// ---------------------------------------------------------------------------
// Kernel 2: causal flash attention. One wave per (batch, 16-row tile).
// ---------------------------------------------------------------------------
__global__ void __launch_bounds__(128, 1)
attn_kernel(const __bf16* __restrict__ qb, const __bf16* __restrict__ kb,
            const __bf16* __restrict__ vT, float* __restrict__ out) {
  __shared__ __bf16 psm[4][16 * 32];           // per-wave P tile (16 x 32)
  const int lane = threadIdx.x & 31;
  const int wv   = threadIdx.x >> 5;
  const int wg   = blockIdx.x * 4 + wv;        // 4096 waves total
  const int b    = wg >> 4;
  const int m0   = (wg & 15) << 4;
  const int l15  = lane & 15;
  const int h    = lane >> 4;
  __bf16* P = psm[wv];

  // Q fragments (A-matrix 16x64 -> two 16x32 fragments), kept in registers
  const __bf16* qrow = qb + (size_t)(b * T_ + m0 + l15) * DK_;
  v16bf qf0 = mk16(ld8(qrow + 16 * h),      ld8(qrow + 16 * h + 8));
  v16bf qf1 = mk16(ld8(qrow + 32 + 16 * h), ld8(qrow + 32 + 16 * h + 8));

  v8f O[4];
#pragma unroll
  for (int i = 0; i < 4; ++i) O[i] = zero8();
  float mrun[8], lrun[8];
#pragma unroll
  for (int r = 0; r < 8; ++r) { mrun[r] = NEG_BIG; lrun[r] = 0.f; }

  const int mlast = m0 + 15;
  const __bf16* vbat = vT + (size_t)b * DV_ * T_;

  for (int n0 = 0; n0 <= mlast; n0 += 32) {    // 32-key chunks (wave-uniform)
#pragma unroll
    for (int sub = 0; sub < 2; ++sub) {
      const int s0 = n0 + 16 * sub;
      if (s0 <= mlast) {                       // wave-uniform branch
        // ---- S tile = Q[16x64] * K^T[64x16]; both K frags loaded first ----
        const __bf16* krow = kb + (size_t)(b * T_ + s0 + l15) * DK_;
        v16bf kf0 = mk16(ld8(krow + 16 * h),      ld8(krow + 16 * h + 8));
        v16bf kf1 = mk16(ld8(krow + 32 + 16 * h), ld8(krow + 32 + 16 * h + 8));
        v8f s = zero8();
        s = wmma_bf16(qf0, kf0, s);
        s = wmma_bf16(qf1, kf1, s);
        __builtin_amdgcn_sched_group_barrier(SG_VMEMR, 4, 0); // K loads
        __builtin_amdgcn_sched_group_barrier(SG_WMMA,  2, 0); // S wmmas
        // ---- scale + causal mask + online softmax ----
        const int col = s0 + l15;
#pragma unroll
        for (int r = 0; r < 8; ++r) {
          const int row = m0 + r + 8 * h;
          float sv = (col <= row) ? s[r] * 0.125f : NEG_BIG;  // 1/sqrt(64)
          float mx = sv;
#pragma unroll
          for (int off = 8; off >= 1; off >>= 1)
            mx = fmaxf(mx, __shfl_xor(mx, off, 32));
          const float newm = fmaxf(mrun[r], mx);
          const float corr = __expf(mrun[r] - newm);
          const float pe   = __expf(sv - newm);
          float rs = pe;
#pragma unroll
          for (int off = 8; off >= 1; off >>= 1)
            rs += __shfl_xor(rs, off, 32);
          lrun[r] = lrun[r] * corr + rs;
          mrun[r] = newm;
#pragma unroll
          for (int vt = 0; vt < 4; ++vt) O[vt][r] *= corr;
          P[(r + 8 * h) * 32 + 16 * sub + l15] = f2bf(pe);
        }
      } else {
        // trailing odd half-chunk: zero the unused P half (exp(-inf)=0)
#pragma unroll
        for (int r = 0; r < 8; ++r)
          P[(r + 8 * h) * 32 + 16 * sub + l15] = f2bf(0.0f);
      }
    }
    // ---- O += P[16x32] * V[32x64]; all four V frags loaded first ----
    const __bf16* prow = P + l15 * 32;
    v16bf pa = mk16(*(const v8bf*)(prow + 16 * h),
                    *(const v8bf*)(prow + 16 * h + 8));
    const __bf16* vr0 = vbat + (size_t)(l15)      * T_ + n0 + 16 * h;
    const __bf16* vr1 = vbat + (size_t)(16 + l15) * T_ + n0 + 16 * h;
    const __bf16* vr2 = vbat + (size_t)(32 + l15) * T_ + n0 + 16 * h;
    const __bf16* vr3 = vbat + (size_t)(48 + l15) * T_ + n0 + 16 * h;
    v16bf vf0 = mk16(ld8(vr0), ld8(vr0 + 8));
    v16bf vf1 = mk16(ld8(vr1), ld8(vr1 + 8));
    v16bf vf2 = mk16(ld8(vr2), ld8(vr2 + 8));
    v16bf vf3 = mk16(ld8(vr3), ld8(vr3 + 8));
    O[0] = wmma_bf16(pa, vf0, O[0]);
    O[1] = wmma_bf16(pa, vf1, O[1]);
    O[2] = wmma_bf16(pa, vf2, O[2]);
    O[3] = wmma_bf16(pa, vf3, O[3]);
    __builtin_amdgcn_sched_group_barrier(SG_DSR,   2, 0); // P reload
    __builtin_amdgcn_sched_group_barrier(SG_VMEMR, 8, 0); // V loads
    __builtin_amdgcn_sched_group_barrier(SG_WMMA,  4, 0); // PV wmmas
  }

  // ---- epilogue: O / l, store f32 ----
#pragma unroll
  for (int r = 0; r < 8; ++r) {
    const float inv = 1.0f / lrun[r];
    const int t = m0 + r + 8 * h;
#pragma unroll
    for (int vt = 0; vt < 4; ++vt)
      out[(size_t)(b * T_ + t) * DV_ + vt * 16 + l15] = O[vt][r] * inv;
  }
}

// ---------------------------------------------------------------------------
// Launch: ws layout = Wt bf16 (147456 B) | q bf16 (8 MB) | k bf16 (8 MB) |
// vT bf16 (8 MB)  -> ~24.2 MB total scratch.
// ---------------------------------------------------------------------------
extern "C" void kernel_launch(void* const* d_in, const int* in_sizes, int n_in,
                              void* d_out, int out_size, void* d_ws,
                              size_t ws_size, hipStream_t stream) {
  (void)in_sizes; (void)n_in; (void)out_size; (void)ws_size;
  const float* x  = (const float*)d_in[0];
  const float* Wq = (const float*)d_in[1];
  const float* Wk = (const float*)d_in[2];
  const float* Wv = (const float*)d_in[3];
  float* out = (float*)d_out;

  char* ws = (char*)d_ws;
  __bf16* Wt = (__bf16*)ws;
  __bf16* qb = (__bf16*)(ws + (size_t)NCOLS * DM_ * 2);       // 147456 B
  __bf16* kb = qb + (size_t)B_ * T_ * DK_;
  __bf16* vT = kb + (size_t)B_ * T_ * DK_;

  wt_pack_kernel<<<(NCOLS * DM_ + 255) / 256, 256, 0, stream>>>(Wq, Wk, Wv, Wt);
  // 4096 tiles, 8 waves (256 threads) per block
  qkv_proj_kernel<<<512, 256, 0, stream>>>(x, Wt, qb, kb, vT);
  // 4096 waves, 4 waves (128 threads) per block
  attn_kernel<<<1024, 128, 0, stream>>>(qb, kb, vT, out);
}
